// MultiHeadAttention_22110491640608
// MI455X (gfx1250) — compile-verified, matrix operand order checked by reading, not run on hardware
//
#include <hip/hip_runtime.h>
#include <hip/hip_bf16.h>

typedef __bf16 bf16;
typedef bf16  v8bf  __attribute__((ext_vector_type(8)));
typedef bf16  v16bf __attribute__((ext_vector_type(16)));
typedef float v8f   __attribute__((ext_vector_type(8)));
typedef unsigned int u32x4 __attribute__((ext_vector_type(4)));
typedef int          i32x4 __attribute__((ext_vector_type(4)));
typedef int          i32x8 __attribute__((ext_vector_type(8)));

constexpr int CB  = 4;
constexpr int CT  = 2048;
constexpr int CD  = 2048;
constexpr int CH  = 16;
constexpr int CHD = 128;

__device__ __forceinline__ v16bf pack_frag(const bf16* p) {
    // A/B fragment: 8 bf16 at p, 8 bf16 at p+16 (K-chunks per ISA 16-bit layout)
    v8bf lo = *(const v8bf*)p;
    v8bf hi = *(const v8bf*)(p + 16);
    v16bf r;
#pragma unroll
    for (int i = 0; i < 8; ++i) { r[i] = lo[i]; r[i + 8] = hi[i]; }
    return r;
}

__device__ __forceinline__ v8f wmma_bf16(v16bf a, v16bf b, v8f c) {
    return __builtin_amdgcn_wmma_f32_16x16x32_bf16(false, a, false, b, (short)0, c,
                                                   false, false);
}

// Build TDM descriptor groups (ISA 8.3/8.4) for a 2D bf16 tile load to LDS.
// tensor: dim0 (contiguous) x dim1, row stride = stride0 elements.
// tile:   tile0 (contiguous) x tile1, LDS padded by (pad_amt_code+1) DWORDs
//         every 2^(pad_int_code+1) DWORDs.
__device__ __forceinline__ void tdm_load_2d(unsigned lds_byte_addr,
                                            const void* gaddr, int dim0, int dim1,
                                            int tile0, int tile1, int stride0,
                                            int pad_int_code, int pad_amt_code) {
    const unsigned long long ga = (unsigned long long)(uintptr_t)gaddr;
    u32x4 g0;
    g0[0] = 1u;                                   // count=1, user descriptor
    g0[1] = lds_byte_addr;                        // lds_addr
    g0[2] = (unsigned)ga;                         // global_addr[31:0]
    g0[3] = (unsigned)((ga >> 32) & 0x1FFFFFFu)   // global_addr[56:32]
            | 0x80000000u;                        // type=2 (bits 127:126)
    i32x8 g1;
    g1[0] = (1 << 16)                             // data_size = 2 bytes
            | (1 << 20)                           // pad_enable
            | (pad_int_code << 22)                // pad_interval
            | (pad_amt_code << 25);               // pad_amount
    g1[1] = (dim0 & 0xFFFF) << 16;                // tensor_dim0[15:0]
    g1[2] = ((dim0 >> 16) & 0xFFFF) | ((dim1 & 0xFFFF) << 16);
    g1[3] = ((dim1 >> 16) & 0xFFFF) | ((tile0 & 0xFFFF) << 16);
    g1[4] = (tile1 & 0xFFFF);                     // tile_dim1, tile_dim2=0
    g1[5] = stride0;                              // tensor_dim0_stride[31:0]
    g1[6] = 0;
    g1[7] = 0;
    i32x4 z4 = {0, 0, 0, 0};
    i32x8 z8 = {0, 0, 0, 0, 0, 0, 0, 0};
    __builtin_amdgcn_tensor_load_to_lds(g0, g1, z4, z4, z8, 0);
}

// ---------------------------------------------------------------------------
// MODE 0: qkv projection. A = x (fp32, MxK), W = w_qkv (fp32, KxN=6144).
//         Epilogue scatters bf16 -> Q/K [B,H,T,HD] (Q scaled 1/HD) and
//         V transposed [B,H,HD,T] (so attention V staging is a pure 2D copy).
// MODE 1: out projection. A = attn output (bf16, MxK), W = w_out (fp32),
//         fp32 epilogue to Cout.
// Block 256 threads (8 waves), tile 128x128, K-step 32, LDS double-buffered.
// ---------------------------------------------------------------------------
template <int MODE>
__global__ __launch_bounds__(256) void gemm_wmma(const void* __restrict__ Ap,
                                                 const float* __restrict__ W,
                                                 bf16* __restrict__ Qb,
                                                 bf16* __restrict__ Kb,
                                                 bf16* __restrict__ Vb,
                                                 float* __restrict__ Cout,
                                                 int M, int N, int K) {
    __shared__ bf16 sA[2][128][40];  // [m][k], pad 32->40 (80B rows, 16B aligned)
    __shared__ bf16 sB[2][128][40];  // transposed: [n][k]

    const int tid  = threadIdx.x;
    const int lane = tid & 31;
    const int wid  = tid >> 5;       // 0..7
    const int wr   = wid & 3;        // wave row group (4 x 32 rows)
    const int wc   = wid >> 2;       // wave col group (2 x 64 cols)
    const int lr   = lane & 15;
    const int kh   = (lane >> 4) * 8;
    const int m0   = blockIdx.y * 128;
    const int n0   = blockIdx.x * 128;

    v8f acc[2][4];
#pragma unroll
    for (int mi = 0; mi < 2; ++mi)
#pragma unroll
        for (int ni = 0; ni < 4; ++ni) acc[mi][ni] = {};

    const int arow = tid >> 1;        // 0..127
    const int akb  = (tid & 1) * 16;  // 0 / 16
    const int bkk  = tid >> 3;        // 0..31
    const int bnb  = (tid & 7) * 16;  // 0..112

    auto load_tile = [&](int k0, int bsel) {
        if constexpr (MODE == 0) {
            const float* ag = (const float*)Ap + (size_t)(m0 + arow) * K + k0 + akb;
#pragma unroll
            for (int i = 0; i < 16; i += 4) {
                float4 f = *(const float4*)(ag + i);
                sA[bsel][arow][akb + i + 0] = (bf16)f.x;
                sA[bsel][arow][akb + i + 1] = (bf16)f.y;
                sA[bsel][arow][akb + i + 2] = (bf16)f.z;
                sA[bsel][arow][akb + i + 3] = (bf16)f.w;
            }
        } else {
            const bf16* ag = (const bf16*)Ap + (size_t)(m0 + arow) * K + k0 + akb;
            *(v8bf*)&sA[bsel][arow][akb]     = *(const v8bf*)(ag);
            *(v8bf*)&sA[bsel][arow][akb + 8] = *(const v8bf*)(ag + 8);
        }
        const float* wg = W + (size_t)(k0 + bkk) * N + n0 + bnb;
#pragma unroll
        for (int i = 0; i < 16; ++i) sB[bsel][bnb + i][bkk] = (bf16)wg[i];
    };

    const int nk = K >> 5;
    load_tile(0, 0);
    __syncthreads();

    for (int kt = 0; kt < nk; ++kt) {
        const int cur = kt & 1;
        if (kt + 1 < nk) load_tile((kt + 1) << 5, cur ^ 1);

        v16bf af[2], bfv[4];
#pragma unroll
        for (int mi = 0; mi < 2; ++mi)
            af[mi] = pack_frag(&sA[cur][wr * 32 + mi * 16 + lr][kh]);
#pragma unroll
        for (int ni = 0; ni < 4; ++ni)
            bfv[ni] = pack_frag(&sB[cur][wc * 64 + ni * 16 + lr][kh]);
#pragma unroll
        for (int mi = 0; mi < 2; ++mi)
#pragma unroll
            for (int ni = 0; ni < 4; ++ni)
                acc[mi][ni] = wmma_bf16(af[mi], bfv[ni], acc[mi][ni]);
        __syncthreads();
    }

    // ---- epilogue: C layout per ISA (VGPR r -> M=r / M=8+r, N = lane&15)
    const int hi8 = kh;
#pragma unroll
    for (int mi = 0; mi < 2; ++mi) {
#pragma unroll
        for (int ni = 0; ni < 4; ++ni) {
#pragma unroll
            for (int r = 0; r < 8; ++r) {
                const int m = m0 + wr * 32 + mi * 16 + hi8 + r;
                const int n = n0 + wc * 64 + ni * 16 + lr;
                const float v = acc[mi][ni][r];
                if constexpr (MODE == 0) {
                    const int which = n >> 11;  // 0=q 1=k 2=v
                    const int d     = n & 2047;
                    const int h     = d >> 7;
                    const int hd    = d & 127;
                    const int b     = m >> 11;  // T = 2048
                    const int t     = m & 2047;
                    if (which == 0)
                        Qb[((size_t)(b * CH + h) * CT + t) * CHD + hd] =
                            (bf16)(v * (1.0f / 128.0f));  // both hd^-0.5 folded
                    else if (which == 1)
                        Kb[((size_t)(b * CH + h) * CT + t) * CHD + hd] = (bf16)v;
                    else  // V stored transposed: [B,H,HD,T]
                        Vb[((size_t)(b * CH + h) * CHD + hd) * CT + t] = (bf16)v;
                } else {
                    Cout[(size_t)m * N + n] = v;
                }
            }
        }
    }
}

// ---------------------------------------------------------------------------
// In-place RoPE on Q and K (bf16, [B,H,T,HD]). One thread per (bh,t,j<64)
// pair -> both elements of the rotation pair handled by one thread (no race).
// ---------------------------------------------------------------------------
__global__ __launch_bounds__(256) void rope_kernel(bf16* __restrict__ Qb,
                                                   bf16* __restrict__ Kb,
                                                   const float* __restrict__ sinT,
                                                   const float* __restrict__ cosT) {
    const int idx = blockIdx.x * blockDim.x + threadIdx.x;  // B*H*T*64
    const int j   = idx & 63;
    const int t   = (idx >> 6) & (CT - 1);
    const int bh  = idx >> 17;
    const size_t base = ((size_t)bh * CT + t) * CHD;
    const float c1 = cosT[t * CHD + j],      s1 = sinT[t * CHD + j];
    const float c2 = cosT[t * CHD + j + 64], s2 = sinT[t * CHD + j + 64];

    float a1 = (float)Qb[base + j], a2 = (float)Qb[base + j + 64];
    Qb[base + j]      = (bf16)(a1 * c1 - a2 * s1);
    Qb[base + j + 64] = (bf16)(a2 * c2 + a1 * s2);

    a1 = (float)Kb[base + j]; a2 = (float)Kb[base + j + 64];
    Kb[base + j]      = (bf16)(a1 * c1 - a2 * s1);
    Kb[base + j + 64] = (bf16)(a2 * c2 + a1 * s2);
}

// ---------------------------------------------------------------------------
// Flash-style causal attention. Block = 128 threads (4 waves), 64 q-rows per
// block (16 per wave, resident as A-fragments). kv tiles of 64 are DMAed by
// the Tensor Data Mover (wave 0 issues, s_wait_tensorcnt + barrier publishes),
// double-buffered in LDS with TDM hardware row padding (+16B) so fragment
// reads are conflict-free ds_load_b128. V comes from the transposed [B,H,HD,T]
// workspace, making its tile a pure 2D copy.
// ---------------------------------------------------------------------------
__global__ __launch_bounds__(128) void attn_kernel(const bf16* __restrict__ Qb,
                                                   const bf16* __restrict__ Kb,
                                                   const bf16* __restrict__ Vt,
                                                   bf16* __restrict__ Ob) {
    __shared__ bf16 sK[2][64][136];   // [kv][hd], 256B rows + 16B TDM pad
    __shared__ bf16 sVt[2][128][72];  // [hd][kv], 128B rows + 16B TDM pad
    __shared__ bf16 sP[4][16][72];    // per-wave P staging

    const int tid  = threadIdx.x;
    const int lane = tid & 31;
    const int wid  = tid >> 5;  // 0..3
    const int lr   = lane & 15;
    const int kh   = (lane >> 4) * 8;
    const int q0   = blockIdx.x * 64;
    const int bh   = blockIdx.y;

    const bf16* Qg  = Qb + (size_t)bh * CT * CHD;
    const bf16* Kg  = Kb + (size_t)bh * CT * CHD;
    const bf16* Vtg = Vt + (size_t)bh * CHD * CT;

    auto issue_tiles = [&](int kv0, int bsel) {
        // K tile: tensor 128x2048 (dim0=HD contiguous), tile 128x64,
        // pad 4 DWORDs every 64 DWORDs -> LDS row stride 272B.
        tdm_load_2d((unsigned)(uintptr_t)&sK[bsel][0][0],
                    Kg + (size_t)kv0 * CHD, CHD, CT, CHD, 64, CHD,
                    /*interval 2^6=64dw*/ 5, /*amount 4dw*/ 3);
        // V tile: tensor 2048x128 (dim0=T contiguous), tile 64x128,
        // pad 4 DWORDs every 32 DWORDs -> LDS row stride 144B.
        tdm_load_2d((unsigned)(uintptr_t)&sVt[bsel][0][0],
                    Vtg + kv0, CT, CHD, 64, CHD, CT,
                    /*interval 2^5=32dw*/ 4, /*amount 4dw*/ 3);
    };

    // resident Q fragments: 16 rows x 128 (4 K-steps)
    v16bf qf[4];
    {
        const bf16* qr = Qg + (size_t)(q0 + wid * 16 + lr) * CHD;
#pragma unroll
        for (int s = 0; s < 4; ++s) qf[s] = pack_frag(qr + s * 32 + kh);
    }

    v8f oacc[8];
#pragma unroll
    for (int oi = 0; oi < 8; ++oi) oacc[oi] = {};
    float mrow[8], lsum[8];
#pragma unroll
    for (int r = 0; r < 8; ++r) { mrow[r] = -3.0e38f; lsum[r] = 0.0f; }

    const int ntile = (q0 >> 6) + 1;
    if (wid == 0) issue_tiles(0, 0);

    for (int it = 0; it < ntile; ++it) {
        const int kv0 = it << 6;
        const int cur = it & 1;
        if (wid == 0) {
            if (it + 1 < ntile) {
                issue_tiles((it + 1) << 6, cur ^ 1);
                __builtin_amdgcn_s_wait_tensorcnt(2);  // current pair done
            } else {
                __builtin_amdgcn_s_wait_tensorcnt(0);
            }
        }
        __syncthreads();  // publish TDM-filled tiles to all waves

        // ---- S = Q @ K^T  (16 x 64 per wave, fp32 accum)
        v8f sfr[4];
#pragma unroll
        for (int nf = 0; nf < 4; ++nf) sfr[nf] = {};
#pragma unroll
        for (int nf = 0; nf < 4; ++nf)
#pragma unroll
            for (int s = 0; s < 4; ++s)
                sfr[nf] = wmma_bf16(
                    qf[s], pack_frag(&sK[cur][nf * 16 + lr][s * 32 + kh]), sfr[nf]);

        // ---- causal mask + online softmax (rows private to 16-lane halves)
        const int qbase = q0 + wid * 16 + kh;
#pragma unroll
        for (int r = 0; r < 8; ++r) {
            const int mg = qbase + r;
            float mx = -3.0e38f;
#pragma unroll
            for (int nf = 0; nf < 4; ++nf) {
                const int cg = kv0 + nf * 16 + lr;
                if (cg > mg) sfr[nf][r] = -3.0e38f;
                mx = fmaxf(mx, sfr[nf][r]);
            }
#pragma unroll
            for (int off = 1; off < 16; off <<= 1)
                mx = fmaxf(mx, __shfl_xor(mx, off, 16));
            const float mnew = fmaxf(mrow[r], mx);
            const float corr = __expf(mrow[r] - mnew);
            mrow[r] = mnew;
            float ps = 0.0f;
#pragma unroll
            for (int nf = 0; nf < 4; ++nf) {
                const float p = __expf(sfr[nf][r] - mnew);
                sfr[nf][r] = p;
                ps += p;
            }
#pragma unroll
            for (int off = 1; off < 16; off <<= 1) ps += __shfl_xor(ps, off, 16);
            lsum[r] = lsum[r] * corr + ps;
#pragma unroll
            for (int oi = 0; oi < 8; ++oi) oacc[oi][r] *= corr;
        }

        // ---- P: C-layout -> LDS -> A-layout (per-wave region, in-order DS)
#pragma unroll
        for (int nf = 0; nf < 4; ++nf)
#pragma unroll
            for (int r = 0; r < 8; ++r)
                sP[wid][kh + r][nf * 16 + lr] = (bf16)sfr[nf][r];
        asm volatile("s_wait_dscnt 0" ::: "memory");

        // ---- O += P @ V
#pragma unroll
        for (int s2 = 0; s2 < 2; ++s2) {
            v16bf pf = pack_frag(&sP[wid][lr][s2 * 32 + kh]);
#pragma unroll
            for (int oi = 0; oi < 8; ++oi)
                oacc[oi] = wmma_bf16(
                    pf, pack_frag(&sVt[cur][oi * 16 + lr][s2 * 32 + kh]), oacc[oi]);
        }
        __syncthreads();  // tile consumed; next iteration may overwrite it
    }

    // ---- epilogue: normalize, write [B,T,D] bf16
    const int b = bh >> 4;
    const int h = bh & 15;
#pragma unroll
    for (int oi = 0; oi < 8; ++oi) {
#pragma unroll
        for (int r = 0; r < 8; ++r) {
            const int tq = q0 + wid * 16 + kh + r;
            const int hd = oi * 16 + lr;
            Ob[((size_t)(b * CT + tq)) * CD + h * CHD + hd] =
                (bf16)(oacc[oi][r] / lsum[r]);
        }
    }
}

// ---------------------------------------------------------------------------
extern "C" void kernel_launch(void* const* d_in, const int* in_sizes, int n_in,
                              void* d_out, int out_size, void* d_ws, size_t ws_size,
                              hipStream_t stream) {
    (void)in_sizes; (void)n_in; (void)out_size; (void)ws_size;
    const float* x     = (const float*)d_in[0];
    const float* w_qkv = (const float*)d_in[1];
    const float* w_out = (const float*)d_in[2];
    const float* sinT  = (const float*)d_in[3];
    const float* cosT  = (const float*)d_in[4];
    float* out = (float*)d_out;

    // workspace: Q,K [B,H,T,HD] bf16; V [B,H,HD,T] bf16; O [B,T,D] bf16.
    char* ws = (char*)d_ws;
    const size_t SEG = (size_t)CB * CH * CT * CHD * sizeof(bf16);  // 32 MiB
    bf16* Qb = (bf16*)(ws);
    bf16* Kb = (bf16*)(ws + SEG);
    bf16* Vb = (bf16*)(ws + 2 * SEG);
    bf16* Ob = (bf16*)(ws + 3 * SEG);

    const int M = CB * CT;  // 8192

    gemm_wmma<0><<<dim3((3 * CD) / 128, M / 128), 256, 0, stream>>>(
        (const void*)x, w_qkv, Qb, Kb, Vb, nullptr, M, 3 * CD, CD);

    rope_kernel<<<(CB * CH * CT * 64) / 256, 256, 0, stream>>>(Qb, Kb, sinT, cosT);

    attn_kernel<<<dim3(CT / 64, CB * CH), 128, 0, stream>>>(Qb, Kb, Vb, Ob);

    gemm_wmma<1><<<dim3(CD / 128, M / 128), 256, 0, stream>>>(
        (const void*)Ob, w_out, nullptr, nullptr, nullptr, out, M, CD, CD);
}
